// MultiHeadAttention_20444044329771
// MI455X (gfx1250) — compile-verified, matrix operand order checked by reading, not run on hardware
//
#include <hip/hip_runtime.h>
#include <math.h>

typedef __attribute__((ext_vector_type(16))) _Float16 v16h;
typedef __attribute__((ext_vector_type(8)))  _Float16 v8h;
typedef __attribute__((ext_vector_type(8)))  float    v8f;
typedef __attribute__((ext_vector_type(4)))  int      v4i;

#define D_MODEL 1024
#define NUM_HEADS 16
#define DEPTH 64
#define BATCH 2
#define SEQ 2048
#define MTOT (BATCH * SEQ)   // 4096

// CDNA5 async global->LDS copies (ASYNCcnt path). Guarded so either toolchain
// compiles; host pass always takes the plain fallback.
#if defined(__HIP_DEVICE_COMPILE__) &&                                   \
    __has_builtin(__builtin_amdgcn_global_load_async_to_lds_b128) &&     \
    __has_builtin(__builtin_amdgcn_s_wait_asynccnt)
#define USE_ASYNC_LDS 1
#else
#define USE_ASYNC_LDS 0
#endif

#if USE_ASYNC_LDS
typedef __attribute__((address_space(1))) v4i* as1_v4i_ptr;
typedef __attribute__((address_space(3))) v4i* as3_v4i_ptr;
#endif

__device__ __forceinline__ void copy16_g2l(void* lds, const void* g) {
#if USE_ASYNC_LDS
  __builtin_amdgcn_global_load_async_to_lds_b128(
      (as1_v4i_ptr)(void*)g, (as3_v4i_ptr)lds, 0, 0);
#else
  *(v8h*)lds = *(const v8h*)g;
#endif
}

__device__ __forceinline__ void async_fence() {
#if USE_ASYNC_LDS
  __builtin_amdgcn_s_wait_asynccnt(0);
#endif
}

struct h4 { _Float16 x, y, z, w; };

// ---------------------------------------------------------------------------
// fp32 -> f16 conversion (vectorized)
// ---------------------------------------------------------------------------
__global__ void cvt_kernel(const float* __restrict__ in, _Float16* __restrict__ out, int n) {
  int i = (blockIdx.x * blockDim.x + threadIdx.x) * 4;
  if (i < n) {
    float4 v = *(const float4*)(in + i);
    h4 o;
    o.x = (_Float16)v.x; o.y = (_Float16)v.y;
    o.z = (_Float16)v.z; o.w = (_Float16)v.w;
    *(h4*)(out + i) = o;
  }
}

// ---------------------------------------------------------------------------
// Fragment loader for 16-bit A/B operands of v_wmma_*_16x16x32.
// Lanes 0-15 -> row lo, K {0..7, 16..23}; lanes 16-31 -> row lo, K {8..15, 24..31}.
// Caller passes lane-resolved pointer (base + lo*stride + hi*8), 16B aligned.
// ---------------------------------------------------------------------------
__device__ __forceinline__ v16h ld_frag(const _Float16* p) {
  v8h a = *(const v8h*)p;
  v8h b = *(const v8h*)(p + 16);
  v16h r;
#pragma unroll
  for (int i = 0; i < 8; ++i) { r[i] = a[i]; r[i + 8] = b[i]; }
  return r;
}

// ---------------------------------------------------------------------------
// GEMM: Y[M,1024] = X[M,1024] @ W[1024,1024] + bias
// Block: 256 threads (8 waves), tile 128x128, BK=32.
// Wave (wm,wn) in 4x2 grid computes 32x64 via 2x4 WMMA tiles.
// ---------------------------------------------------------------------------
template <bool OUT_F32>
__global__ __launch_bounds__(256)
void gemm_f16_kernel(const _Float16* __restrict__ X, const _Float16* __restrict__ W,
                     const float* __restrict__ bias, float* __restrict__ Yf,
                     _Float16* __restrict__ Yh) {
  __shared__ __align__(16) _Float16 As[128 * 32];  // m-major, k-contiguous
  __shared__ __align__(16) _Float16 Bt[128 * 32];  // n-major, k-contiguous (transposed W tile)

  const int t    = threadIdx.x;
  const int lane = t & 31;
  const int wid  = t >> 5;
  const int lo   = lane & 15;
  const int hi   = lane >> 4;
  const int wm   = wid >> 1;  // 0..3
  const int wn   = wid & 1;   // 0..1
  const int m0   = blockIdx.y * 128;
  const int n0   = blockIdx.x * 128;

  const v8f zero = {0, 0, 0, 0, 0, 0, 0, 0};
  v8f acc[2][4];
#pragma unroll
  for (int i = 0; i < 2; ++i)
#pragma unroll
    for (int j = 0; j < 4; ++j) acc[i][j] = zero;

  for (int k0 = 0; k0 < 1024; k0 += 32) {
    // stage A (async DMA into LDS): 128x32 f16 = 512 chunks of 8, 2 per thread
#pragma unroll
    for (int i = 0; i < 2; ++i) {
      int c   = t + i * 256;
      int row = c >> 2;
      int c8  = (c & 3) * 8;
      copy16_g2l(&As[row * 32 + c8],
                 &X[(size_t)(m0 + row) * 1024 + k0 + c8]);
    }
    // stage W tile transposed (VGPR path; async cannot transpose)
#pragma unroll
    for (int i = 0; i < 2; ++i) {
      int c    = t + i * 256;
      int krow = c >> 4;
      int nc   = (c & 15) * 8;
      v8h wv = *(const v8h*)&W[(size_t)(k0 + krow) * 1024 + n0 + nc];
#pragma unroll
      for (int j = 0; j < 8; ++j) Bt[(nc + j) * 32 + krow] = wv[j];
    }
    // prefetch next iteration's tiles into cache while we compute
    if (k0 + 32 < 1024) {
      int c = t;
      __builtin_prefetch(&X[(size_t)(m0 + (c >> 2)) * 1024 + k0 + 32 + (c & 3) * 8], 0, 1);
      __builtin_prefetch(&W[(size_t)(k0 + 32 + (c >> 4)) * 1024 + n0 + (c & 15) * 8], 0, 1);
    }
    async_fence();
    __syncthreads();

    v16h af[2], bf[4];
#pragma unroll
    for (int mt = 0; mt < 2; ++mt)
      af[mt] = ld_frag(&As[(wm * 32 + mt * 16 + lo) * 32 + hi * 8]);
#pragma unroll
    for (int nt = 0; nt < 4; ++nt)
      bf[nt] = ld_frag(&Bt[(wn * 64 + nt * 16 + lo) * 32 + hi * 8]);
#pragma unroll
    for (int mt = 0; mt < 2; ++mt)
#pragma unroll
      for (int nt = 0; nt < 4; ++nt)
        acc[mt][nt] = __builtin_amdgcn_wmma_f32_16x16x32_f16(
            false, af[mt], false, bf[nt], (short)0, acc[mt][nt], false, false);
    __syncthreads();
  }

  // epilogue: C layout row M = r + 8*hi, col N = lo (within tile)
#pragma unroll
  for (int mt = 0; mt < 2; ++mt)
#pragma unroll
    for (int nt = 0; nt < 4; ++nt) {
      int   col = n0 + wn * 64 + nt * 16 + lo;
      float bv  = bias[col];
#pragma unroll
      for (int r = 0; r < 8; ++r) {
        int   rowg = m0 + wm * 32 + mt * 16 + r + hi * 8;
        float v    = acc[mt][nt][r] + bv;
        if (OUT_F32) Yf[(size_t)rowg * 1024 + col] = v;
        else         Yh[(size_t)rowg * 1024 + col] = (_Float16)v;
      }
    }
}

// ---------------------------------------------------------------------------
// Flash attention: one block per (64-query tile, b*h). 4 waves, 16 rows/wave.
// Q/K/V are f16 in [b, s, h*64+d] layout. ctx written in same layout.
// ---------------------------------------------------------------------------
__global__ __launch_bounds__(128)
void attention_kernel(const _Float16* __restrict__ Qh, const _Float16* __restrict__ Kh,
                      const _Float16* __restrict__ Vh, const float* __restrict__ mask,
                      _Float16* __restrict__ ctx) {
  __shared__ __align__(16) _Float16 Qs[64 * 64];       // q-row major, d contiguous
  __shared__ __align__(16) _Float16 Ks[64 * 64];       // key major, d contiguous
  __shared__ __align__(16) _Float16 Vt[64 * 64];       // d major, key contiguous (transposed)
  __shared__ __align__(16) _Float16 Ps[4 * 16 * 64];   // per-wave P tile, k contiguous

  const int t    = threadIdx.x;
  const int lane = t & 31;
  const int wid  = t >> 5;
  const int lo   = lane & 15;
  const int hi   = lane >> 4;
  const int bh   = blockIdx.y;
  const int b    = bh >> 4;
  const int h    = bh & 15;
  const int s0   = blockIdx.x * 64;
  const size_t base = (size_t)b * SEQ * 1024 + (size_t)h * 64;

  // stage Q tile (64x64 f16) via async DMA: 512 chunks of 8, 4 per thread
#pragma unroll
  for (int i = 0; i < 4; ++i) {
    int c = t + i * 128;
    int row = c >> 3, d0 = (c & 7) * 8;
    copy16_g2l(&Qs[row * 64 + d0],
               &Qh[base + (size_t)(s0 + row) * 1024 + d0]);
  }

  const v8f zero = {0, 0, 0, 0, 0, 0, 0, 0};
  float mrow[8], lrow[8];
  v8f acco[4];
#pragma unroll
  for (int r = 0; r < 8; ++r) { mrow[r] = -3.0e38f; lrow[r] = 0.0f; }
#pragma unroll
  for (int dt = 0; dt < 4; ++dt) acco[dt] = zero;

  for (int kb = 0; kb < SEQ; kb += 64) {
    __syncthreads();  // protect Ks/Vt/Ps from previous iteration's readers
    // stage K (async) and V transposed (VGPR path)
#pragma unroll
    for (int i = 0; i < 4; ++i) {
      int c = t + i * 128;
      int row = c >> 3, d0 = (c & 7) * 8;
      copy16_g2l(&Ks[row * 64 + d0],
                 &Kh[base + (size_t)(kb + row) * 1024 + d0]);
      v8h vv = *(const v8h*)&Vh[base + (size_t)(kb + row) * 1024 + d0];
#pragma unroll
      for (int j = 0; j < 8; ++j) Vt[(d0 + j) * 64 + row] = vv[j];
    }
    // prefetch next key block while this one is being consumed
    if (kb + 64 < SEQ) {
      int c = t;
      int row = c >> 3, d0 = (c & 7) * 8;
      __builtin_prefetch(&Kh[base + (size_t)(kb + 64 + row) * 1024 + d0], 0, 1);
      __builtin_prefetch(&Vh[base + (size_t)(kb + 64 + row) * 1024 + d0], 0, 1);
    }
    async_fence();
    __syncthreads();

    // S = Q @ K^T : wave computes 16 rows x 64 keys (4 N-tiles), Kdim = depth 64
    v8f sacc[4];
#pragma unroll
    for (int nt = 0; nt < 4; ++nt) sacc[nt] = zero;
#pragma unroll
    for (int kk = 0; kk < 64; kk += 32) {
      v16h aq = ld_frag(&Qs[(wid * 16 + lo) * 64 + kk + hi * 8]);
#pragma unroll
      for (int nt = 0; nt < 4; ++nt) {
        v16h bk = ld_frag(&Ks[(nt * 16 + lo) * 64 + kk + hi * 8]);
        sacc[nt] = __builtin_amdgcn_wmma_f32_16x16x32_f16(
            false, aq, false, bk, (short)0, sacc[nt], false, false);
      }
    }

    // scaled logits + mask, then online softmax. C layout: row r+8*hi, col lo.
    float p[4][8];
    float mb[8];
#pragma unroll
    for (int r = 0; r < 8; ++r) mb[r] = -3.0e38f;
#pragma unroll
    for (int nt = 0; nt < 4; ++nt) {
      int kg = kb + nt * 16 + lo;
#pragma unroll
      for (int r = 0; r < 8; ++r) {
        int   qg = s0 + wid * 16 + r + hi * 8;
        float v  = sacc[nt][r] * 0.125f + mask[(size_t)qg * SEQ + kg] * (-1e9f);
        p[nt][r] = v;
        mb[r]    = fmaxf(mb[r], v);
      }
    }
    // row max across the 16 lanes of each half (stays in C layout)
#pragma unroll
    for (int m = 1; m < 16; m <<= 1)
#pragma unroll
      for (int r = 0; r < 8; ++r) mb[r] = fmaxf(mb[r], __shfl_xor(mb[r], m, 32));

    float sums[8];
#pragma unroll
    for (int r = 0; r < 8; ++r) {
      float mn   = fmaxf(mrow[r], mb[r]);
      float corr = __expf(mrow[r] - mn);
      mrow[r]    = mn;
      float s = 0.0f;
#pragma unroll
      for (int nt = 0; nt < 4; ++nt) {
        p[nt][r] = __expf(p[nt][r] - mn);
        s += p[nt][r];
      }
      sums[r] = s;
      lrow[r] = lrow[r] * corr;
#pragma unroll
      for (int dt = 0; dt < 4; ++dt) acco[dt][r] = acco[dt][r] * corr;
    }
#pragma unroll
    for (int m = 1; m < 16; m <<= 1)
#pragma unroll
      for (int r = 0; r < 8; ++r) sums[r] += __shfl_xor(sums[r], m, 32);
#pragma unroll
    for (int r = 0; r < 8; ++r) lrow[r] += sums[r];

    // spill P (C layout) to LDS as A-operand feedstock (row-major, k-contig)
#pragma unroll
    for (int nt = 0; nt < 4; ++nt)
#pragma unroll
      for (int r = 0; r < 8; ++r)
        Ps[wid * 1024 + (r + hi * 8) * 64 + nt * 16 + lo] = (_Float16)p[nt][r];
    __syncthreads();

    // O += P @ V : Kdim = 64 keys, 4 depth tiles
#pragma unroll
    for (int kk = 0; kk < 64; kk += 32) {
      v16h ap = ld_frag(&Ps[wid * 1024 + lo * 64 + kk + hi * 8]);
#pragma unroll
      for (int dt = 0; dt < 4; ++dt) {
        v16h bv = ld_frag(&Vt[(dt * 16 + lo) * 64 + kk + hi * 8]);
        acco[dt] = __builtin_amdgcn_wmma_f32_16x16x32_f16(
            false, ap, false, bv, (short)0, acco[dt], false, false);
      }
    }
  }

  // normalize and write context (f16, same [b,s,h*64+d] layout)
#pragma unroll
  for (int dt = 0; dt < 4; ++dt)
#pragma unroll
    for (int r = 0; r < 8; ++r) {
      int   qg = s0 + wid * 16 + r + hi * 8;
      float o  = acco[dt][r] / lrow[r];
      ctx[base + (size_t)qg * 1024 + dt * 16 + lo] = (_Float16)o;
    }
}

// ---------------------------------------------------------------------------
extern "C" void kernel_launch(void* const* d_in, const int* in_sizes, int n_in,
                              void* d_out, int out_size, void* d_ws, size_t ws_size,
                              hipStream_t stream) {
  (void)in_sizes; (void)n_in; (void)out_size; (void)ws_size;

  const float* q    = (const float*)d_in[0];
  const float* k    = (const float*)d_in[1];
  const float* v    = (const float*)d_in[2];
  const float* mask = (const float*)d_in[3];
  const float* wq   = (const float*)d_in[4];
  const float* bq   = (const float*)d_in[5];
  const float* wk   = (const float*)d_in[6];
  const float* bk   = (const float*)d_in[7];
  const float* wv   = (const float*)d_in[8];
  const float* bv   = (const float*)d_in[9];
  const float* wo   = (const float*)d_in[10];
  const float* bo   = (const float*)d_in[11];
  float* out = (float*)d_out;

  const size_t SZX = (size_t)MTOT * 1024;     // 4,194,304
  const size_t SZW = (size_t)1024 * 1024;     // 1,048,576
  _Float16* w16 = (_Float16*)d_ws;            // total 64 MiB of f16 scratch
  _Float16* qx  = w16;
  _Float16* kx  = qx + SZX;
  _Float16* vx  = kx + SZX;
  _Float16* wqh = vx + SZX;
  _Float16* wkh = wqh + SZW;
  _Float16* wvh = wkh + SZW;
  _Float16* woh = wvh + SZW;
  _Float16* Qp  = woh + SZW;
  _Float16* Kp  = Qp + SZX;
  _Float16* Vp  = Kp + SZX;
  _Float16* Cx  = Vp + SZX;

  const int cb = 256;
  cvt_kernel<<<dim3((unsigned)(SZX / 4 / cb)), cb, 0, stream>>>(q, qx, (int)SZX);
  cvt_kernel<<<dim3((unsigned)(SZX / 4 / cb)), cb, 0, stream>>>(k, kx, (int)SZX);
  cvt_kernel<<<dim3((unsigned)(SZX / 4 / cb)), cb, 0, stream>>>(v, vx, (int)SZX);
  cvt_kernel<<<dim3((unsigned)(SZW / 4 / cb)), cb, 0, stream>>>(wq, wqh, (int)SZW);
  cvt_kernel<<<dim3((unsigned)(SZW / 4 / cb)), cb, 0, stream>>>(wk, wkh, (int)SZW);
  cvt_kernel<<<dim3((unsigned)(SZW / 4 / cb)), cb, 0, stream>>>(wv, wvh, (int)SZW);
  cvt_kernel<<<dim3((unsigned)(SZW / 4 / cb)), cb, 0, stream>>>(wo, woh, (int)SZW);

  dim3 gg(1024 / 128, MTOT / 128);  // (8, 32)
  gemm_f16_kernel<false><<<gg, 256, 0, stream>>>(qx, wqh, bq, nullptr, Qp);
  gemm_f16_kernel<false><<<gg, 256, 0, stream>>>(kx, wkh, bk, nullptr, Kp);
  gemm_f16_kernel<false><<<gg, 256, 0, stream>>>(vx, wvh, bv, nullptr, Vp);

  attention_kernel<<<dim3(SEQ / 64, BATCH * NUM_HEADS), 128, 0, stream>>>(
      Qp, Kp, Vp, mask, Cx);

  gemm_f16_kernel<true><<<gg, 256, 0, stream>>>(Cx, woh, bo, out, nullptr);
}
